// MSAMambaBlock_28406913696269
// MI455X (gfx1250) — matrix operand order repaired and checked by reading, not compiled, not run on hardware
//
#include <hip/hip_runtime.h>

// ---------------------------------------------------------------- constants
#define B_    4
#define M_    32
#define L_    1024
#define D_    256
#define DI    512
#define DS    16
#define DTR   16
#define DCONV 4
#define H_    4
#define DATT  256
#define DMEM  1024
#define NSEQ  (B_ * M_)        // 128
#define NTOK  (NSEQ * L_)      // 131072
#define EPS_  1e-5f

// WMMA tiling
#define BM 64
#define BN 64
#define KC 32

typedef __attribute__((ext_vector_type(16))) __bf16        v16bf;
typedef __attribute__((ext_vector_type(8)))  float         v8f;
typedef __attribute__((ext_vector_type(4)))  unsigned int  v4u;
typedef __attribute__((ext_vector_type(8)))  int           v8i;
typedef __attribute__((ext_vector_type(4)))  int           v4i;

__device__ __forceinline__ float sigmoid_(float x) { return 1.0f / (1.0f + __expf(-x)); }
__device__ __forceinline__ float softplus_(float x) { return (x > 20.0f) ? x : log1pf(__expf(x)); }

// ---------------------------------------------------------------- TDM support
#if defined(__has_builtin)
#  if __has_builtin(__builtin_amdgcn_tensor_load_to_lds) && __has_builtin(__builtin_amdgcn_s_wait_tensorcnt)
#    define HAVE_TDM 1
#  endif
#endif
#ifndef HAVE_TDM
#  define HAVE_TDM 0
#endif

#if HAVE_TDM
#warning "CDNA5 probe: TDM path ENABLED (tensor_load_to_lds + s_wait_tensorcnt)"
// 2D tile load: global (height x width f32 rows, row stride in elems) -> LDS (packed)
// Descriptor per CDNA5 ISA.
__device__ __forceinline__ void tdm_load_2d(unsigned lds_addr, unsigned long long gaddr,
                                            unsigned width, unsigned height,
                                            unsigned long long row_stride_elems) {
    v4u g0;
    g0[0] = 1u;                                                  // count=1, user mode
    g0[1] = lds_addr;                                            // LDS byte address
    g0[2] = (unsigned)(gaddr & 0xffffffffu);                     // global_addr[31:0]
    g0[3] = (unsigned)((gaddr >> 32) & 0x1ffffffu) | (2u << 30); // global_addr[56:32] | type=2
    v8i g1;
    g1[0] = (int)(2u << 16);                                     // data_size=4B, no multicast
    g1[1] = (int)((width & 0xffffu) << 16);                      // tensor_dim0[15:0] in [63:48]
    g1[2] = (int)(((width >> 16) & 0xffffu) |                    // tensor_dim0[31:16]
                  ((height & 0xffffu) << 16));                   // tensor_dim1[15:0]
    g1[3] = (int)(((height >> 16) & 0xffffu) |                   // tensor_dim1[31:16]
                  ((width & 0xffffu) << 16));                    // tile_dim0 = width
    g1[4] = (int)(height & 0xffffu);                             // tile_dim1 = height, tile_dim2=0
    g1[5] = (int)(row_stride_elems & 0xffffffffull);             // tensor_dim0_stride[31:0]
    g1[6] = (int)((row_stride_elems >> 32) & 0xffffull);         // tensor_dim0_stride[47:32]
    g1[7] = 0;
    v4i z4 = {0, 0, 0, 0};
#if __clang_major__ >= 23
    v8i z8 = {0, 0, 0, 0, 0, 0, 0, 0};
    __builtin_amdgcn_tensor_load_to_lds(g0, g1, z4, z4, z8, 0);
#else
    __builtin_amdgcn_tensor_load_to_lds(g0, g1, z4, z4, 0);
#endif
}
#else
#warning "CDNA5 probe: TDM builtins NOT available - attention uses manual LDS staging"
#endif

// ---------------------------------------------------------------- row inv-RMS
__global__ void rowscale_kernel(const float* __restrict__ X, float* __restrict__ s, int nrows) {
    int gwave = (int)((blockIdx.x * blockDim.x + threadIdx.x) >> 5);
    int lane  = threadIdx.x & 31;
    if (gwave >= nrows) return;
    const float* row = X + (size_t)gwave * D_;
    float acc = 0.0f;
    #pragma unroll
    for (int i = lane; i < D_; i += 32) { float t = row[i]; acc += t * t; }
    #pragma unroll
    for (int off = 16; off; off >>= 1) acc += __shfl_xor(acc, off, 32);
    if (lane == 0) s[gwave] = rsqrtf(acc * (1.0f / (float)D_) + EPS_);
}

__global__ void finalnorm_kernel(const float* __restrict__ X, const float* __restrict__ w,
                                 float* __restrict__ out, int nrows) {
    int gwave = (int)((blockIdx.x * blockDim.x + threadIdx.x) >> 5);
    int lane  = threadIdx.x & 31;
    if (gwave >= nrows) return;
    const float* row = X + (size_t)gwave * D_;
    float acc = 0.0f;
    #pragma unroll
    for (int i = lane; i < D_; i += 32) { float t = row[i]; acc += t * t; }
    #pragma unroll
    for (int off = 16; off; off >>= 1) acc += __shfl_xor(acc, off, 32);
    float inv = rsqrtf(acc * (1.0f / (float)D_) + EPS_);
    #pragma unroll
    for (int i = lane; i < D_; i += 32) out[(size_t)gwave * D_ + i] = row[i] * w[i] * inv;
}

// ---------------------------------------------------------------- WMMA GEMM
// out[nrows,ncols] = A[nrows,K(lda)] @ W[ncols,K]^T (+ epilogue)
// nrows % 64 == 0 always. BOUNDS=false additionally requires K % KC == 0 and ncols % BN == 0.
// ASCALE: A-load multiplies aweight[col]*ascale[row] (fused rmsnorm)
// EPI 0: plain   1: softplus(acc+bias)   2: silu(acc+bias)
// EPI 3: acc (+bias) + R[*rweight*rscale if RSCALE]
template <int EPI, bool ASCALE, bool RSCALE, bool BOUNDS>
__global__ void __launch_bounds__(256)
gemm_bf16_wmma(const float* __restrict__ A, int lda,
               const float* __restrict__ ascale, const float* __restrict__ aweight,
               const float* __restrict__ W, const float* __restrict__ bias,
               const float* __restrict__ R, const float* __restrict__ rscale,
               const float* __restrict__ rweight,
               float* __restrict__ out, int nrows, int ncols, int K) {
    __shared__ __bf16 As[2][BM][KC + 8];
    __shared__ __bf16 Bs[2][KC][BN + 8];

    const int tid   = threadIdx.x;
    const int wvid  = tid >> 5;
    const int lane  = tid & 31;
    const int waveM = wvid >> 1;
    const int waveN = wvid & 1;
    const int half  = lane >> 4;
    const int l16   = lane & 15;
    const int rowBase = blockIdx.y * BM;
    const int colBase = blockIdx.x * BN;

    // stage one 64x32 A tile + 32x64 B tile into LDS buffer `buf` (f32 -> bf16)
    auto stage = [&](int buf, int k0) {
        #pragma unroll
        for (int i = 0; i < 2; ++i) {
            int idx = i * 256 + tid;          // 0..511
            int r   = idx >> 3;               // row 0..63
            int qc  = (idx & 7) << 2;         // col 0,4,...,28
            int gr  = rowBase + r;
            int gk  = k0 + qc;
            float4 val;
            if (!BOUNDS || (gk + 3 < K)) {
                val = *(const float4*)(A + (size_t)gr * lda + gk);
            } else {
                val = make_float4(0.f, 0.f, 0.f, 0.f);
                float* vp = (float*)&val;
                #pragma unroll
                for (int j = 0; j < 4; ++j)
                    if (gk + j < K) vp[j] = A[(size_t)gr * lda + gk + j];
            }
            if (ASCALE) {
                float  rs = ascale[gr];
                float4 aw = *(const float4*)(aweight + gk);
                val.x *= aw.x * rs; val.y *= aw.y * rs;
                val.z *= aw.z * rs; val.w *= aw.w * rs;
            }
            As[buf][r][qc + 0] = (__bf16)val.x;
            As[buf][r][qc + 1] = (__bf16)val.y;
            As[buf][r][qc + 2] = (__bf16)val.z;
            As[buf][r][qc + 3] = (__bf16)val.w;
        }
        #pragma unroll
        for (int i = 0; i < 2; ++i) {
            int idx = i * 256 + tid;
            int n   = idx >> 3;               // 0..63
            int kq  = (idx & 7) << 2;         // 0,4,...,28
            int gn  = colBase + n;
            int gk  = k0 + kq;
            float4 val;
            if (!BOUNDS) {
                val = *(const float4*)(W + (size_t)gn * K + gk);
            } else {
                val = make_float4(0.f, 0.f, 0.f, 0.f);
                if (gn < ncols) {
                    if (gk + 3 < K) {
                        val = *(const float4*)(W + (size_t)gn * K + gk);
                    } else {
                        float* vp = (float*)&val;
                        #pragma unroll
                        for (int j = 0; j < 4; ++j)
                            if (gk + j < K) vp[j] = W[(size_t)gn * K + gk + j];
                    }
                }
            }
            Bs[buf][kq + 0][n] = (__bf16)val.x;
            Bs[buf][kq + 1][n] = (__bf16)val.y;
            Bs[buf][kq + 2][n] = (__bf16)val.z;
            Bs[buf][kq + 3][n] = (__bf16)val.w;
        }
    };

    v8f acc0 = {};
    v8f acc1 = {};

    stage(0, 0);
    int cur = 0;
    for (int k0 = 0; k0 < K; k0 += KC) {
        int nxt = k0 + KC;
        __syncthreads();                       // staging of `cur` complete; prior reads drained
        if (nxt < K) stage(cur ^ 1, nxt);      // overlap next stage with this tile's WMMAs
        if (nxt + KC < K) {                    // prefetch tile after next
            __builtin_prefetch(A + (size_t)(rowBase + (tid & 63)) * lda + (nxt + KC), 0, 1);
            __builtin_prefetch(W + (size_t)(colBase + (tid & 63)) * K + (nxt + KC), 0, 1);
        }

        // A fragment: row m = waveM*16 + l16; K striped per CDNA5 16-bit A layout
        v16bf af;
        #pragma unroll
        for (int e = 0; e < 16; ++e) {
            int kk = (e < 8) ? (half * 8 + e) : (16 + half * 8 + (e - 8));
            af[e] = As[cur][waveM * 16 + l16][kk];
        }
        // B fragments: n = l16, K striped same as A
        v16bf b0, b1;
        #pragma unroll
        for (int e = 0; e < 16; ++e) {
            int kk = (e < 8) ? (half * 8 + e) : (16 + half * 8 + (e - 8));
            b0[e] = Bs[cur][kk][waveN * 32 + l16];
            b1[e] = Bs[cur][kk][waveN * 32 + 16 + l16];
        }
        acc0 = __builtin_amdgcn_wmma_f32_16x16x32_bf16(false, af, false, b0, (short)0, acc0, false, false);
        acc1 = __builtin_amdgcn_wmma_f32_16x16x32_bf16(false, af, false, b1, (short)0, acc1, false, false);
        cur ^= 1;
    }

    // ---- epilogue: m = waveM*16 + half*8 + r, n = waveN*32 + t*16 + l16
    #pragma unroll
    for (int t = 0; t < 2; ++t) {
        v8f accv = t ? acc1 : acc0;
        int gc = colBase + waveN * 32 + t * 16 + l16;
        if (BOUNDS && gc >= ncols) continue;
        #pragma unroll
        for (int r = 0; r < 8; ++r) {
            int gr = rowBase + waveM * 16 + half * 8 + r;
            float v = accv[r];
            if (EPI == 1) { v += bias[gc]; v = softplus_(v); }
            else if (EPI == 2) { v += bias[gc]; v = v * sigmoid_(v); }
            else if (EPI == 3) {
                if (bias) v += bias[gc];
                float rv = R[(size_t)gr * ncols + gc];
                if (RSCALE) rv *= rweight[gc] * rscale[gr];
                v += rv;
            }
            out[(size_t)gr * ncols + gc] = v;
        }
    }
}

// ---------------------------------------------------------------- depthwise conv + SiLU
__global__ void conv_silu_kernel(const float* __restrict__ xz, const float* __restrict__ cw,
                                 const float* __restrict__ cb, float* __restrict__ xc) {
    size_t idx = (size_t)blockIdx.x * blockDim.x + threadIdx.x;   // NTOK*DI
    if (idx >= (size_t)NTOK * DI) return;
    int    c     = (int)(idx & (DI - 1));
    size_t tok   = idx / DI;
    int    l     = (int)(tok & (L_ - 1));
    size_t sbase = tok - l;
    float acc = cb[c];
    #pragma unroll
    for (int kk = 0; kk < DCONV; ++kk) {
        int lp = l + kk - (DCONV - 1);
        if (lp >= 0) acc += cw[c * DCONV + kk] * xz[(sbase + lp) * (size_t)(2 * DI) + c];
    }
    xc[idx] = acc * sigmoid_(acc);
}

// ---------------------------------------------------------------- selective scan
__global__ void scan_kernel(const float* __restrict__ delta, const float* __restrict__ xc,
                            const float* __restrict__ xdbl, const float* __restrict__ xz,
                            const float* __restrict__ A_log, const float* __restrict__ Dv,
                            float* __restrict__ yz) {
    int idx = blockIdx.x * blockDim.x + threadIdx.x;   // NSEQ*DI
    if (idx >= NSEQ * DI) return;
    int c = idx & (DI - 1);
    int s = idx / DI;
    float Ar[DS], h[DS];
    #pragma unroll
    for (int j = 0; j < DS; ++j) { Ar[j] = -__expf(A_log[c * DS + j]); h[j] = 0.0f; }
    float Dc = Dv[c];
    for (int l = 0; l < L_; ++l) {
        size_t row = (size_t)s * L_ + l;
        float dlt = delta[row * DI + c];
        float u   = xc[row * DI + c];
        float du  = dlt * u;
        float bc[2 * DS];                      // B then C, b128 loads
        const float4* p4 = (const float4*)(xdbl + row * (size_t)(DTR + 2 * DS) + DTR);
        #pragma unroll
        for (int j = 0; j < (2 * DS) / 4; ++j) ((float4*)bc)[j] = p4[j];
        float y = 0.0f;
        #pragma unroll
        for (int j = 0; j < DS; ++j) {
            h[j] = __expf(dlt * Ar[j]) * h[j] + du * bc[j];
            y += h[j] * bc[DS + j];
        }
        float z = xz[row * (size_t)(2 * DI) + DI + c];
        yz[row * DI + c] = (y + Dc * u) * (z * sigmoid_(z));
    }
}

// ---------------------------------------------------------------- attention over M (32x32, hd=64)
// one wave per (b,l,h) instance; K/V staged into LDS by the Tensor Data Mover
__global__ void __launch_bounds__(64)
attention_kernel(const float* __restrict__ q, const float* __restrict__ k,
                 const float* __restrict__ v, float* __restrict__ att) {
    __shared__ float kbuf[2][M_][64];
    __shared__ float vbuf[2][M_][64];
    int wvid = threadIdx.x >> 5;
    int lane = threadIdx.x & 31;
    int inst = blockIdx.x * 2 + wvid;       // B_*L_*H_ = 16384 total
    int b    = inst >> 12;                  // / (L_*H_)
    int rem  = inst & 4095;
    int l    = rem >> 2;
    int hh   = rem & 3;

    size_t base_elem = ((size_t)(b * M_) * L_ + l) * DATT + (size_t)hh * 64;
#if HAVE_TDM
    {
        typedef __attribute__((address_space(3))) float lds_f;
        unsigned lk = (unsigned)(unsigned long long)(lds_f*)&kbuf[wvid][0][0];
        unsigned lv = (unsigned)(unsigned long long)(lds_f*)&vbuf[wvid][0][0];
        unsigned long long gk = (unsigned long long)(const void*)k + base_elem * 4ull;
        unsigned long long gv = (unsigned long long)(const void*)v + base_elem * 4ull;
        tdm_load_2d(lk, gk, 64u, (unsigned)M_, (unsigned long long)L_ * DATT);
        tdm_load_2d(lv, gv, 64u, (unsigned)M_, (unsigned long long)L_ * DATT);
        __builtin_amdgcn_s_wait_tensorcnt(0);
    }
#else
    for (int idx = lane; idx < M_ * 64; idx += 32) {
        int m = idx >> 6, e = idx & 63;
        size_t row = (size_t)(b * M_ + m) * L_ + l;
        kbuf[wvid][m][e] = k[row * DATT + hh * 64 + e];
        vbuf[wvid][m][e] = v[row * DATT + hh * 64 + e];
    }
#endif
    __syncthreads();

    int m = lane;
    size_t rowm = (size_t)(b * M_ + m) * L_ + l;
    float qr[64];
    #pragma unroll
    for (int e = 0; e < 64; ++e) qr[e] = q[rowm * DATT + hh * 64 + e];

    float sc[M_];
    float mx = -1e30f;
    for (int n = 0; n < M_; ++n) {
        float sdot = 0.0f;
        #pragma unroll
        for (int e = 0; e < 64; ++e) sdot += qr[e] * kbuf[wvid][n][e];
        sdot *= 0.125f;                      // 1/sqrt(64)
        sc[n] = sdot;
        mx = fmaxf(mx, sdot);
    }
    float ssum = 0.0f;
    for (int n = 0; n < M_; ++n) { sc[n] = __expf(sc[n] - mx); ssum += sc[n]; }
    float inv = 1.0f / ssum;
    for (int e = 0; e < 64; ++e) {
        float a = 0.0f;
        for (int n = 0; n < M_; ++n) a += sc[n] * vbuf[wvid][n][e];
        att[rowm * DATT + hh * 64 + e] = a * inv;
    }
}

// ---------------------------------------------------------------- launcher
extern "C" void kernel_launch(void* const* d_in, const int* in_sizes, int n_in,
                              void* d_out, int out_size, void* d_ws, size_t ws_size,
                              hipStream_t stream) {
    (void)in_sizes; (void)n_in; (void)out_size; (void)ws_size;
    const float* x         = (const float*)d_in[0];
    const float* in_proj_w = (const float*)d_in[1];
    const float* conv_w    = (const float*)d_in[2];
    const float* conv_b    = (const float*)d_in[3];
    const float* x_proj_w  = (const float*)d_in[4];
    const float* dt_proj_w = (const float*)d_in[5];
    const float* dt_proj_b = (const float*)d_in[6];
    const float* A_log     = (const float*)d_in[7];
    const float* Dv        = (const float*)d_in[8];
    const float* out_proj_w= (const float*)d_in[9];
    const float* mamba_w   = (const float*)d_in[10];
    const float* wq        = (const float*)d_in[11];
    const float* wk        = (const float*)d_in[12];
    const float* wv        = (const float*)d_in[13];
    const float* wo        = (const float*)d_in[14];
    const float* w1        = (const float*)d_in[15];
    const float* b1        = (const float*)d_in[16];
    const float* w2        = (const float*)d_in[17];
    const float* b2        = (const float*)d_in[18];
    const float* norm1_w   = (const float*)d_in[19];
    const float* norm2_w   = (const float*)d_in[20];
    const float* norm3_w   = (const float*)d_in[21];
    float* out = (float*)d_out;

    // workspace carve (aliased across phases)
    float* ws = (float*)d_ws;
    size_t off = 0;
    const size_t NT = (size_t)NTOK;
    float* fA = ws + off; off += NT * 1024;  // xz -> att(256) -> mlp_h(1024)
    float* fB = ws + off; off += NT * 512;   // xc -> q(256)
    float* fC = ws + off; off += NT * 512;   // delta -> k(256)
    float* fD = ws + off; off += NT * 512;   // yz -> v(256)
    float* fE = ws + off; off += NT * 48;    // xdbl
    float* fF = ws + off; off += NT * 256;   // res1 -> res3
    float* fG = ws + off; off += NT * 256;   // res2
    float* s0 = ws + off; off += NT;         // invrms(mamba norm)
    float* s1 = ws + off; off += NT;         // invrms(norm1)
    float* s2 = ws + off; off += NT;         // invrms(norm2)

    dim3 blk(256);
    const int rsGrid = NTOK / 8;
    auto gg = [](int ncols, int nrows) { return dim3((ncols + BN - 1) / BN, (nrows + BM - 1) / BM); };

    // ---- Mamba path
    rowscale_kernel<<<rsGrid, blk, 0, stream>>>(x, s0, NTOK);
    gemm_bf16_wmma<0, true, false, false><<<gg(2 * DI, NTOK), blk, 0, stream>>>( // xz = rmsnorm(x) @ in_proj^T
        x, D_, s0, mamba_w, in_proj_w, nullptr, nullptr, nullptr, nullptr,
        fA, NTOK, 2 * DI, D_);
    conv_silu_kernel<<<(int)((NT * DI) / 256), blk, 0, stream>>>(fA, conv_w, conv_b, fB);
    gemm_bf16_wmma<0, false, false, true><<<gg(DTR + 2 * DS, NTOK), blk, 0, stream>>>( // xdbl = xc @ x_proj^T
        fB, DI, nullptr, nullptr, x_proj_w, nullptr, nullptr, nullptr, nullptr,
        fE, NTOK, DTR + 2 * DS, DI);
    gemm_bf16_wmma<1, false, false, true><<<gg(DI, NTOK), blk, 0, stream>>>( // delta = softplus(dt @ dt_proj^T + b)
        fE, DTR + 2 * DS, nullptr, nullptr, dt_proj_w, dt_proj_b, nullptr, nullptr, nullptr,
        fC, NTOK, DI, DTR);
    scan_kernel<<<(NSEQ * DI) / 256, blk, 0, stream>>>(fC, fB, fE, fA, A_log, Dv, fD);
    gemm_bf16_wmma<3, false, false, false><<<gg(D_, NTOK), blk, 0, stream>>>( // res1 = yz @ out_proj^T + x
        fD, DI, nullptr, nullptr, out_proj_w, nullptr, x, nullptr, nullptr,
        fF, NTOK, D_, DI);

    // ---- Attention path (o1 = rmsnorm(res1) fused via s1/norm1_w)
    rowscale_kernel<<<rsGrid, blk, 0, stream>>>(fF, s1, NTOK);
    gemm_bf16_wmma<0, true, false, false><<<gg(DATT, NTOK), blk, 0, stream>>>(
        fF, D_, s1, norm1_w, wq, nullptr, nullptr, nullptr, nullptr, fB, NTOK, DATT, D_);
    gemm_bf16_wmma<0, true, false, false><<<gg(DATT, NTOK), blk, 0, stream>>>(
        fF, D_, s1, norm1_w, wk, nullptr, nullptr, nullptr, nullptr, fC, NTOK, DATT, D_);
    gemm_bf16_wmma<0, true, false, false><<<gg(DATT, NTOK), blk, 0, stream>>>(
        fF, D_, s1, norm1_w, wv, nullptr, nullptr, nullptr, nullptr, fD, NTOK, DATT, D_);
    attention_kernel<<<(B_ * L_ * H_) / 2, dim3(64), 0, stream>>>(fB, fC, fD, fA);
    gemm_bf16_wmma<3, false, true, false><<<gg(D_, NTOK), blk, 0, stream>>>( // res2 = att @ wo^T + o1
        fA, DATT, nullptr, nullptr, wo, nullptr, fF, s1, norm1_w,
        fG, NTOK, D_, DATT);

    // ---- MLP path (o2 = rmsnorm(res2) fused via s2/norm2_w)
    rowscale_kernel<<<rsGrid, blk, 0, stream>>>(fG, s2, NTOK);
    gemm_bf16_wmma<2, true, false, false><<<gg(DMEM, NTOK), blk, 0, stream>>>( // mlp_h = silu(o2 @ w1^T + b1)
        fG, D_, s2, norm2_w, w1, b1, nullptr, nullptr, nullptr,
        fA, NTOK, DMEM, D_);
    gemm_bf16_wmma<3, false, true, false><<<gg(D_, NTOK), blk, 0, stream>>>( // res3 = mlp_h @ w2^T + b2 + o2
        fA, DMEM, nullptr, nullptr, w2, b2, fG, s2, norm2_w,
        fF, NTOK, D_, DMEM);
    finalnorm_kernel<<<rsGrid, blk, 0, stream>>>(fF, norm3_w, out, NTOK);
}
